// IsolatedMultiAttention_27547920237189
// MI455X (gfx1250) — compile-verified, matrix operand order checked by reading, not run on hardware
//
#include <hip/hip_runtime.h>

typedef unsigned short u16;
typedef __attribute__((ext_vector_type(16))) __bf16 v16bf;
typedef __attribute__((ext_vector_type(8)))  float  v8f;

union Frag {
  v16bf v;
  uint4 q[2];
};

__device__ __forceinline__ u16 f2bf(float f) {
  unsigned u = __builtin_bit_cast(unsigned, f);
  unsigned r = u + 0x7FFFu + ((u >> 16) & 1u);   // round-to-nearest-even
  return (u16)(r >> 16);
}

__device__ __forceinline__ v8f wmma_bf16(const Frag& a, const Frag& b, v8f c) {
  return __builtin_amdgcn_wmma_f32_16x16x32_bf16(
      false, a.v, false, b.v, (short)0, c, false, false);
}

// ---- CDNA5 async global->LDS copy (ASYNCcnt-tracked, no VGPR data path) ----
__device__ __forceinline__ void async_ld_b128(void* lds_ptr, const void* gptr) {
  uint32_t lds_off = (uint32_t)(uintptr_t)lds_ptr;   // LDS aperture = addr[31:0]
  asm volatile("global_load_async_to_lds_b128 %0, %1, off"
               :: "v"(lds_off), "v"((unsigned long long)(uintptr_t)gptr)
               : "memory");
}
__device__ __forceinline__ void wait_async0() {
  asm volatile("s_wait_asynccnt 0x0" ::: "memory");
}

#define D_MODEL 1024
#define SEQ     2048
#define BATCH   4
#define HEADS   16
#define DK      64
#define MTOT    (BATCH * SEQ)   // 8192

// ---------------- fp32 -> bf16 ----------------
__global__ __launch_bounds__(256) void cvt_f32_to_bf16(
    const float* __restrict__ in, u16* __restrict__ out, int n4) {
  int i = blockIdx.x * blockDim.x + threadIdx.x;
  if (i < n4) {
    float4 f = ((const float4*)in)[i];
    ushort4 o;
    o.x = f2bf(f.x); o.y = f2bf(f.y); o.z = f2bf(f.z); o.w = f2bf(f.w);
    ((ushort4*)out)[i] = o;
  }
}

// ---------------- Y = X @ W^T  (bf16 in, f32 acc) ----------------
// X: [M, K] bf16 row-major, W: [N, K] bf16 row-major.
// Block tile 128x128, K-step 32, double-buffered async LDS staging.
// 8 waves: 4 (M) x 2 (N); each wave computes 32x64 (2x4 WMMA accumulators).
template<bool F32OUT>
__global__ __launch_bounds__(256)
void gemm_x_wT(const u16* __restrict__ X, const u16* __restrict__ W,
               u16* __restrict__ Ybf, float* __restrict__ Yf) {
  constexpr int Kdim = D_MODEL;
  constexpr int Ndim = D_MODEL;
  __shared__ alignas(16) u16 Xs[2][128 * 32];
  __shared__ alignas(16) u16 Ws[2][128 * 32];

  const int tid  = threadIdx.x;
  const int lane = tid & 31;
  const int wave = tid >> 5;
  const int wm   = wave & 3;        // 32-row strip in block tile
  const int wn   = wave >> 2;       // 64-col strip in block tile
  const int mblk = blockIdx.x * 128;
  const int nblk = blockIdx.y * 128;

  const int m     = lane & 15;          // row/col within 16
  const int khalf = (lane >> 4) << 3;   // 0 or 8 (A/B 16-bit lane layout)

  v8f acc[2][4];
#pragma unroll
  for (int i = 0; i < 2; ++i)
#pragma unroll
    for (int j = 0; j < 4; ++j) acc[i][j] = {};

  const int lrow  = tid >> 1;           // 0..127
  const int lkoff = (tid & 1) << 4;     // 0 or 16 elements
  const u16* xg = X + (size_t)(mblk + lrow) * Kdim + lkoff;
  const u16* wg = W + (size_t)(nblk + lrow) * Kdim + lkoff;

  // issue one 128x32 X tile + 128x32 W tile into buffer `buf` via async DMA
  auto issue_tile = [&](int buf, int k0) {
    u16* xs = &Xs[buf][lrow * 32 + lkoff];
    u16* ws = &Ws[buf][lrow * 32 + lkoff];
    async_ld_b128(xs,     xg + k0);
    async_ld_b128(xs + 8, xg + k0 + 8);
    async_ld_b128(ws,     wg + k0);
    async_ld_b128(ws + 8, wg + k0 + 8);
  };

  issue_tile(0, 0);
  wait_async0();
  __syncthreads();

  int cur = 0;
  for (int k0 = 0; k0 < Kdim; k0 += 32) {
    const int nxt = cur ^ 1;
    if (k0 + 32 < Kdim) issue_tile(nxt, k0 + 32);   // prefetch next tile

    Frag a[2], b[4];
#pragma unroll
    for (int i = 0; i < 2; ++i) {
      const u16* p = &Xs[cur][(wm * 32 + i * 16 + m) * 32];
      a[i].q[0] = *(const uint4*)(p + khalf);       // k = khalf..khalf+7
      a[i].q[1] = *(const uint4*)(p + khalf + 16);  // k = 16+khalf..
    }
#pragma unroll
    for (int j = 0; j < 4; ++j) {
      const u16* p = &Ws[cur][(wn * 64 + j * 16 + m) * 32];
      b[j].q[0] = *(const uint4*)(p + khalf);
      b[j].q[1] = *(const uint4*)(p + khalf + 16);
    }
#pragma unroll
    for (int i = 0; i < 2; ++i)
#pragma unroll
      for (int j = 0; j < 4; ++j)
        acc[i][j] = wmma_bf16(a[i], b[j], acc[i][j]);

    wait_async0();      // my async transfers for `nxt` landed in LDS
    __syncthreads();    // everyone's landed; everyone done reading `cur`
    cur = nxt;
  }

  // C layout: VGPR r, lane -> row = r + 8*(lane>=16), col = lane&15
  const int rbase = mblk + wm * 32 + ((lane >> 4) << 3);
  const int cbase = nblk + wn * 64 + m;
#pragma unroll
  for (int i = 0; i < 2; ++i)
#pragma unroll
    for (int j = 0; j < 4; ++j)
#pragma unroll
      for (int r = 0; r < 8; ++r) {
        size_t row = rbase + i * 16 + r;
        size_t col = cbase + j * 16;
        float v = acc[i][j][r];
        if (F32OUT) Yf[row * Ndim + col] = v;
        else        Ybf[row * Ndim + col] = f2bf(v);
      }
}

// ---------------- flash attention, one (b,h) per blockIdx.y ----------------
// 128-query tile per block; 8 waves x 16 query rows; key tiles of 32.
__global__ __launch_bounds__(256)
void attn_flash(const u16* __restrict__ Q, const u16* __restrict__ K,
                const u16* __restrict__ V, u16* __restrict__ Ctx) {
  __shared__ alignas(16) u16 Ks[32 * 64];      // [key][d]  row-major
  __shared__ alignas(16) u16 Vs[64 * 32];      // [d][key]  transposed
  __shared__ alignas(16) u16 Ps[8][16 * 32];   // per-wave P scratch

  const int tid = threadIdx.x, lane = tid & 31, wave = tid >> 5;
  const int bh = blockIdx.y, b = bh >> 4, h = bh & 15;
  const int q0 = blockIdx.x * 128 + wave * 16;
  const int m = lane & 15, khalf = (lane >> 4) << 3;

  const u16* Qb = Q + (size_t)b * SEQ * D_MODEL + h * DK;
  const u16* Kb = K + (size_t)b * SEQ * D_MODEL + h * DK;
  const u16* Vb = V + (size_t)b * SEQ * D_MODEL + h * DK;

  // Q A-fragments (d = 0..31, 32..63), loaded once
  Frag qa[2];
  {
    const u16* qp = Qb + (size_t)(q0 + m) * D_MODEL;
    qa[0].q[0] = *(const uint4*)(qp + khalf);
    qa[0].q[1] = *(const uint4*)(qp + khalf + 16);
    qa[1].q[0] = *(const uint4*)(qp + 32 + khalf);
    qa[1].q[1] = *(const uint4*)(qp + 32 + khalf + 16);
  }

  v8f o[4];
#pragma unroll
  for (int j = 0; j < 4; ++j) o[j] = {};
  float mrow[8], lrow[8];
#pragma unroll
  for (int r = 0; r < 8; ++r) { mrow[r] = -1e30f; lrow[r] = 0.0f; }

  const int ldr = tid >> 3;          // key row 0..31
  const int ldc = (tid & 7) << 3;    // d col  0,8,..,56

  for (int kt = 0; kt < SEQ; kt += 32) {
    // V tile via VGPRs (needs transposed store); K tile via async DMA
    uint4 vv = *(const uint4*)(Vb + (size_t)(kt + ldr) * D_MODEL + ldc);
    __syncthreads();   // all waves done reading previous Ks/Vs
    async_ld_b128(&Ks[ldr * 64 + ldc],
                  Kb + (size_t)(kt + ldr) * D_MODEL + ldc);
    alignas(16) u16 tmp[8];
    *(uint4*)tmp = vv;
#pragma unroll
    for (int i = 0; i < 8; ++i) Vs[(ldc + i) * 32 + ldr] = tmp[i];  // transpose
    wait_async0();
    __syncthreads();

    // scores: S[16q x 32k] = Q(16x64) x K^T, two 16-col WMMA tiles, K chained
    v8f sc0 = {}, sc1 = {};
    {
      Frag kb;
      const u16* p0 = &Ks[m * 64];            // B col n = key n -> K row n
      kb.q[0] = *(const uint4*)(p0 + khalf);
      kb.q[1] = *(const uint4*)(p0 + khalf + 16);
      sc0 = wmma_bf16(qa[0], kb, sc0);
      kb.q[0] = *(const uint4*)(p0 + 32 + khalf);
      kb.q[1] = *(const uint4*)(p0 + 32 + khalf + 16);
      sc0 = wmma_bf16(qa[1], kb, sc0);
      const u16* p1 = &Ks[(16 + m) * 64];
      kb.q[0] = *(const uint4*)(p1 + khalf);
      kb.q[1] = *(const uint4*)(p1 + khalf + 16);
      sc1 = wmma_bf16(qa[0], kb, sc1);
      kb.q[0] = *(const uint4*)(p1 + 32 + khalf);
      kb.q[1] = *(const uint4*)(p1 + 32 + khalf + 16);
      sc1 = wmma_bf16(qa[1], kb, sc1);
    }

    // online softmax: row M lives across a 16-lane group -> shfl_xor(width 16)
    const float scale = 0.125f;   // 1/sqrt(64)
    float alpha[8], p0v[8], p1v[8];
#pragma unroll
    for (int r = 0; r < 8; ++r) {
      float s0 = sc0[r] * scale;
      float s1 = sc1[r] * scale;
      float t = fmaxf(s0, s1);
#pragma unroll
      for (int off = 8; off >= 1; off >>= 1)
        t = fmaxf(t, __shfl_xor(t, off, 16));
      float mn = fmaxf(mrow[r], t);
      alpha[r] = __expf(mrow[r] - mn);
      float p0 = __expf(s0 - mn);
      float p1 = __expf(s1 - mn);
      float ps = p0 + p1;
#pragma unroll
      for (int off = 8; off >= 1; off >>= 1)
        ps += __shfl_xor(ps, off, 16);
      lrow[r] = lrow[r] * alpha[r] + ps;
      mrow[r] = mn;
      p0v[r] = p0; p1v[r] = p1;
    }

    // P (C-layout f32) -> bf16 A-layout via per-wave LDS bounce
    {
      const int rowoff = (lane >> 4) << 3;
#pragma unroll
      for (int r = 0; r < 8; ++r) {
        Ps[wave][(rowoff + r) * 32 + m]      = f2bf(p0v[r]);
        Ps[wave][(rowoff + r) * 32 + 16 + m] = f2bf(p1v[r]);
      }
    }
    // rescale running output by alpha
#pragma unroll
    for (int j = 0; j < 4; ++j)
#pragma unroll
      for (int r = 0; r < 8; ++r) o[j][r] *= alpha[r];

    Frag pa;
    {
      const u16* pp = &Ps[wave][m * 32];
      pa.q[0] = *(const uint4*)(pp + khalf);
      pa.q[1] = *(const uint4*)(pp + khalf + 16);
    }
    // O += P(16x32) x V(32x64): 4 n-tiles from transposed Vs
#pragma unroll
    for (int j = 0; j < 4; ++j) {
      Frag vb;
      const u16* vp = &Vs[(j * 16 + m) * 32];  // B col n = dim n -> Vs row
      vb.q[0] = *(const uint4*)(vp + khalf);
      vb.q[1] = *(const uint4*)(vp + khalf + 16);
      o[j] = wmma_bf16(pa, vb, o[j]);
    }
  }

  // normalize and write context (bf16, feeds final GEMM)
  {
    const int rowoff = (lane >> 4) << 3;
#pragma unroll
    for (int r = 0; r < 8; ++r) {
      float inv = 1.0f / lrow[r];
      size_t row = q0 + rowoff + r;
      u16* cp = Ctx + ((size_t)b * SEQ + row) * D_MODEL + h * DK;
#pragma unroll
      for (int j = 0; j < 4; ++j)
        cp[j * 16 + m] = f2bf(o[j][r] * inv);
    }
  }
}

// ---------------- host entry ----------------
extern "C" void kernel_launch(void* const* d_in, const int* in_sizes, int n_in,
                              void* d_out, int out_size, void* d_ws, size_t ws_size,
                              hipStream_t stream) {
  const float* X  = (const float*)d_in[0];
  const float* Wq = (const float*)d_in[1];
  const float* Wk = (const float*)d_in[2];
  const float* Wv = (const float*)d_in[3];
  const float* Wo = (const float*)d_in[4];
  float* out = (float*)d_out;

  const size_t NX = (size_t)MTOT * D_MODEL;     // 8,388,608
  const size_t NW = (size_t)D_MODEL * D_MODEL;  // 1,048,576
  u16* Xb  = (u16*)d_ws;
  u16* Wqb = Xb + NX;
  u16* Wkb = Wqb + NW;
  u16* Wvb = Wkb + NW;
  u16* Wob = Wvb + NW;
  u16* Qb  = Wob + NW;
  u16* Kb  = Qb + NX;
  u16* Vb  = Kb + NX;
  u16* Cb  = Vb + NX;

  cvt_f32_to_bf16<<<(int)(NX / 4 / 256), 256, 0, stream>>>(X,  Xb,  (int)(NX / 4));
  cvt_f32_to_bf16<<<(int)(NW / 4 / 256), 256, 0, stream>>>(Wq, Wqb, (int)(NW / 4));
  cvt_f32_to_bf16<<<(int)(NW / 4 / 256), 256, 0, stream>>>(Wk, Wkb, (int)(NW / 4));
  cvt_f32_to_bf16<<<(int)(NW / 4 / 256), 256, 0, stream>>>(Wv, Wvb, (int)(NW / 4));
  cvt_f32_to_bf16<<<(int)(NW / 4 / 256), 256, 0, stream>>>(Wo, Wob, (int)(NW / 4));

  dim3 gg(MTOT / 128, D_MODEL / 128);           // 64 x 8
  gemm_x_wT<false><<<gg, 256, 0, stream>>>(Xb, Wqb, Qb, nullptr);
  gemm_x_wT<false><<<gg, 256, 0, stream>>>(Xb, Wkb, Kb, nullptr);
  gemm_x_wT<false><<<gg, 256, 0, stream>>>(Xb, Wvb, Vb, nullptr);

  dim3 ga(SEQ / 128, BATCH * HEADS);            // 16 x 64
  attn_flash<<<ga, 256, 0, stream>>>(Qb, Kb, Vb, Cb);

  gemm_x_wT<true><<<gg, 256, 0, stream>>>(Cb, Wob, nullptr, out);
}
